// Codred_16801912062517
// MI455X (gfx1250) — matrix operand order repaired and checked by reading, not compiled, not run on hardware
//
#include <hip/hip_runtime.h>

// ---------------------------------------------------------------------------
// Problem constants (from the reference)
// ---------------------------------------------------------------------------
#define BAGSZ   8
#define SEQLEN  512
#define DMODEL  768
#define NPDOC   6            // nodes per doc
#define NNODES  48           // BAG * NPD
#define TCELLS  2304         // N*N matrix cells
#define NHEAD   8
#define DHEAD   96
#define DFFN    1024
#define NLAYER  4
#define NREL    97

typedef __attribute__((ext_vector_type(16))) __bf16        bf16x16;
typedef __attribute__((ext_vector_type(8)))  float         f32x8;
typedef __attribute__((ext_vector_type(4)))  unsigned int  u32x4;

union FragAB { bf16x16 v; u32x4 q[2]; unsigned int u[8]; };
union FragC  { f32x8   v; float f[8]; };

__device__ __forceinline__ unsigned short f2bf(float x) {
  unsigned int u = __float_as_uint(x);
  u += 0x7fffu + ((u >> 16) & 1u);          // round-to-nearest-even
  return (unsigned short)(u >> 16);
}

__device__ __forceinline__ float blk_sum(float v, float* red) {
  int t = threadIdx.x;
  red[t] = v; __syncthreads();
  for (int s = 128; s > 0; s >>= 1) { if (t < s) red[t] += red[t + s]; __syncthreads(); }
  float r = red[0]; __syncthreads();
  return r;
}
__device__ __forceinline__ float blk_max(float v, float* red) {
  int t = threadIdx.x;
  red[t] = v; __syncthreads();
  for (int s = 128; s > 0; s >>= 1) { if (t < s) red[t] = fmaxf(red[t], red[t + s]); __syncthreads(); }
  float r = red[0]; __syncthreads();
  return r;
}

// CDNA5 async global->LDS copy (16B per lane), tracked by ASYNCcnt.
// lds_off: byte offset into LDS (low 32 bits of generic shared address).
// base + byte_off: global source (GVS mode: SADDR(64) + VADDR(u32)).
__device__ __forceinline__ void async_copy_b128(unsigned lds_off,
                                                const unsigned short* base,
                                                unsigned byte_off) {
  asm volatile("global_load_async_to_lds_b128 %0, %1, %2"
               :: "v"(lds_off), "v"(byte_off), "s"(base)
               : "memory");
}

// ---------------------------------------------------------------------------
// bf16 WMMA GEMM:  C[M,N] = act(A[M,K] @ B^T + bias),  B given as [N,K] rows.
//   A: bf16 row-major (lda).  B: bf16 [N][K] row-major (ldb).
//   outF : optional f32 output (ldc)
//   outB : optional bf16 output (ldc)
//   outBT: optional bf16 TRANSPOSED output: outBT[col*ldct + row]
// Preconditions:
//   K % 32 == 0; A rows readable for m0..m0+127, B rows readable for
//   n0..n0+127 (callers over-allocate); 16B alignment of row starts.
// Block: 256 threads (8 waves), tile 128x128x32, double-buffered LDS filled
// with global_load_async_to_lds_b128; each wave owns a 64x32 patch ->
// 8 x v_wmma_f32_16x16x32_bf16 per k-step.
// ---------------------------------------------------------------------------
#define BM 128
#define BN 128
#define BKK 32
#define LDSW 40   // padded LDS row stride (halves); 80B keeps 16B alignment

__global__ __launch_bounds__(256) void gemm_bf16_wmma(
    int M, int N, int K,
    const unsigned short* __restrict__ A, int lda,
    const unsigned short* __restrict__ B, int ldb,
    const float* __restrict__ bias,
    float* __restrict__ outF, unsigned short* __restrict__ outB,
    unsigned short* __restrict__ outBT, int ldc, int ldct,
    int relu)
{
  __shared__ __align__(16) unsigned short sA[2][BM * LDSW];
  __shared__ __align__(16) unsigned short sB[2][BN * LDSW];

  const int tid = threadIdx.x, wave = tid >> 5, lane = tid & 31;
  const int m0 = blockIdx.y * BM, n0 = blockIdx.x * BN;
  const int wm = (wave & 1) * 64, wn = (wave >> 1) * 32;

  // staging geometry: thread covers chunk rows r and r+64, 8 halves at col c8
  const int r  = tid >> 2;
  const int c8 = (tid & 3) << 3;

  // issue the 4 per-wave async copies for k-tile k0 into buffer sel
  auto prefetch = [&](int k0, int sel) {
    unsigned aoff0 = ((unsigned)(m0 + r)      * (unsigned)lda + (unsigned)(k0 + c8)) * 2u;
    unsigned aoff1 = ((unsigned)(m0 + r + 64) * (unsigned)lda + (unsigned)(k0 + c8)) * 2u;
    unsigned boff0 = ((unsigned)(n0 + r)      * (unsigned)ldb + (unsigned)(k0 + c8)) * 2u;
    unsigned boff1 = ((unsigned)(n0 + r + 64) * (unsigned)ldb + (unsigned)(k0 + c8)) * 2u;
    async_copy_b128((unsigned)(uintptr_t)&sA[sel][r * LDSW + c8],        A, aoff0);
    async_copy_b128((unsigned)(uintptr_t)&sA[sel][(r + 64) * LDSW + c8], A, aoff1);
    async_copy_b128((unsigned)(uintptr_t)&sB[sel][r * LDSW + c8],        B, boff0);
    async_copy_b128((unsigned)(uintptr_t)&sB[sel][(r + 64) * LDSW + c8], B, boff1);
  };

  FragC acc[4][2];
#pragma unroll
  for (int i = 0; i < 4; ++i)
#pragma unroll
    for (int j = 0; j < 2; ++j)
#pragma unroll
      for (int rr = 0; rr < 8; ++rr) acc[i][j].f[rr] = 0.f;

  // fragment addressing (ISA VGPR layouts)
  // A (16-bit 16x32): lane m=lane&15; halves[0..7]=K kh..kh+7, [8..15]=K 16+kh..
  const int am = lane & 15, kh = (lane >> 4) << 3;
  // B: lane n=lane&15; halves[0..7]=K kb..kb+7, [8..15]=K 16+kb.. (kb = 0 or 8)
  const int bnn = lane & 15, kb = (lane >> 4) << 3;

  const int nk = K / BKK;
  prefetch(0, 0);
  for (int kt = 0; kt < nk; ++kt) {
    const int sel = kt & 1;
    if (kt + 1 < nk) {
      prefetch((kt + 1) * BKK, sel ^ 1);
      asm volatile("s_wait_asynccnt 0x4" ::: "memory");   // tile kt complete
    } else {
      asm volatile("s_wait_asynccnt 0x0" ::: "memory");
    }
    __syncthreads();                                      // all waves' copies landed

    FragAB af[4], bfr[2];
#pragma unroll
    for (int tm = 0; tm < 4; ++tm) {
      const unsigned short* p = &sA[sel][(wm + tm * 16 + am) * LDSW];
      af[tm].q[0] = *reinterpret_cast<const u32x4*>(p + kh);
      af[tm].q[1] = *reinterpret_cast<const u32x4*>(p + 16 + kh);
    }
#pragma unroll
    for (int tn = 0; tn < 2; ++tn) {
      const unsigned short* p = &sB[sel][(wn + tn * 16 + bnn) * LDSW];
      bfr[tn].q[0] = *reinterpret_cast<const u32x4*>(p + kb);
      bfr[tn].q[1] = *reinterpret_cast<const u32x4*>(p + 16 + kb);
    }
#pragma unroll
    for (int tm = 0; tm < 4; ++tm)
#pragma unroll
      for (int tn = 0; tn < 2; ++tn)
        acc[tm][tn].v = __builtin_amdgcn_wmma_f32_16x16x32_bf16(
            false, af[tm].v, false, bfr[tn].v, (short)0, acc[tm][tn].v, false, false);
    __syncthreads();          // all reads of buf[sel] done before it is refilled
  }

  // ---- epilogue: C element rr of lane l -> (M = rr + 8*(l>=16), N = l&15) ----
  const int rbase = m0 + wm + ((lane >> 4) << 3);
  const int cbase = n0 + wn + (lane & 15);
#pragma unroll
  for (int tm = 0; tm < 4; ++tm)
#pragma unroll
    for (int tn = 0; tn < 2; ++tn) {
      int col = cbase + tn * 16;
      if (col >= N) continue;
      float bv = bias ? bias[col] : 0.f;
#pragma unroll
      for (int rr = 0; rr < 8; ++rr) {
        int row = rbase + tm * 16 + rr;
        if (row >= M) continue;
        float val = acc[tm][tn].f[rr] + bv;
        if (relu) val = fmaxf(val, 0.f);
        if (outF)  outF[(size_t)row * ldc + col] = val;
        if (outB)  outB[(size_t)row * ldc + col] = f2bf(val);
        if (outBT) outBT[(size_t)col * ldct + row] = f2bf(val);
      }
    }
}

// ---------------------------------------------------------------------------
// Small helper kernels
// ---------------------------------------------------------------------------
// f32 [K,N] -> bf16 transposed [N,K] (tiled via LDS)
__global__ void k_f32_to_bf16_t(const float* __restrict__ src,
                                unsigned short* __restrict__ dst, int K, int N) {
  __shared__ float tile[32][33];
  int k0 = blockIdx.y * 32, n0 = blockIdx.x * 32;
  int tx = threadIdx.x & 31, ty = threadIdx.x >> 5;     // 32 x 8
  for (int i = ty; i < 32; i += 8) {
    int k = k0 + i, n = n0 + tx;
    tile[i][tx] = (k < K && n < N) ? src[(size_t)k * N + n] : 0.f;
  }
  __syncthreads();
  for (int i = ty; i < 32; i += 8) {
    int n = n0 + i, k = k0 + tx;
    if (n < N && k < K) dst[(size_t)n * K + k] = f2bf(tile[tx][i]);
  }
}

// masked max-pool over each span: 48 blocks, one node each
__global__ void k_span_pool(const float* __restrict__ emb, const int* __restrict__ spans,
                            float* __restrict__ htbF, unsigned short* __restrict__ htbB) {
  int node = blockIdx.x;                     // doc*NPD + nd
  int doc = node / NPDOC;
  int s = spans[node * 2 + 0], e = spans[node * 2 + 1];
  const float* base = emb + (size_t)doc * SEQLEN * DMODEL;
  for (int d = threadIdx.x; d < DMODEL; d += blockDim.x) {
    float m = -3.402823466e38f;
    for (int p = s; p <= e; ++p) m = fmaxf(m, base[(size_t)p * DMODEL + d]);
    htbF[(size_t)node * DMODEL + d] = m;
    htbB[(size_t)node * DMODEL + d] = f2bf(m);
  }
}

// pairA[i*48+j][d] = bf16(relu(u[i][d] + v[j][d]))
__global__ void k_pair(const float* __restrict__ u, const float* __restrict__ v,
                       unsigned short* __restrict__ pairA) {
  int cell = blockIdx.x;
  int i = cell / NNODES, j = cell % NNODES;
  for (int d = threadIdx.x; d < DMODEL; d += blockDim.x) {
    float x = u[(size_t)i * DMODEL + d] + v[(size_t)j * DMODEL + d];
    pairA[(size_t)cell * DMODEL + d] = f2bf(fmaxf(x, 0.f));
  }
}

// row softmax with scale; f32 in -> bf16 probabilities out
__global__ void k_softmax(const float* __restrict__ S, unsigned short* __restrict__ P,
                          int n, float scale) {
  __shared__ float red[256];
  int row = blockIdx.x, tid = threadIdx.x;
  const float* s = S + (size_t)row * n;
  unsigned short* p = P + (size_t)row * n;
  float mx = -3.402823466e38f;
  for (int j = tid; j < n; j += 256) mx = fmaxf(mx, s[j] * scale);
  mx = blk_max(mx, red);
  float sum = 0.f;
  for (int j = tid; j < n; j += 256) sum += __expf(s[j] * scale - mx);
  sum = blk_sum(sum, red);
  float inv = 1.f / sum;
  for (int j = tid; j < n; j += 256) p[j] = f2bf(__expf(s[j] * scale - mx) * inv);
}

// x = LN(x + o) * g + b ; writes f32 + bf16 (in-place safe per-row)
__global__ void k_add_ln(const float* __restrict__ x, const float* __restrict__ o,
                         const float* __restrict__ g, const float* __restrict__ b,
                         float* __restrict__ outF, unsigned short* __restrict__ outB) {
  __shared__ float buf[DMODEL];
  __shared__ float red[256];
  int row = blockIdx.x, tid = threadIdx.x;
  float ls = 0.f, ls2 = 0.f;
  for (int d = tid; d < DMODEL; d += 256) {
    float t = x[(size_t)row * DMODEL + d] + o[(size_t)row * DMODEL + d];
    buf[d] = t; ls += t; ls2 += t * t;
  }
  float mean = blk_sum(ls, red) * (1.f / DMODEL);
  float var  = blk_sum(ls2, red) * (1.f / DMODEL) - mean * mean;
  float rs = rsqrtf(var + 1e-5f);
  __syncthreads();
  for (int d = tid; d < DMODEL; d += 256) {
    float y = (buf[d] - mean) * rs * g[d] + b[d];
    outF[(size_t)row * DMODEL + d] = y;
    outB[(size_t)row * DMODEL + d] = f2bf(y);
  }
}

// gather (h,t) cells -> LN -> @Wp+bp -> ht_logits; bag-max -> bag_logit
// out layout: [0..96]=bag_logit, [97..96+776]=ht_logits (8x97 row-major)
__global__ void k_final(const float* __restrict__ x, const float* __restrict__ ng,
                        const float* __restrict__ nb, const float* __restrict__ Wp,
                        const float* __restrict__ bp, float* __restrict__ out) {
  __shared__ float feat[BAGSZ * DMODEL];
  __shared__ float red[256];
  int tid = threadIdx.x;
  for (int i = 0; i < BAGSZ; ++i) {
    int hp = i * NPDOC;
    int row = hp * NNODES + hp + 1;
    float ls = 0.f, ls2 = 0.f;
    for (int d = tid; d < DMODEL; d += 256) {
      float t = x[(size_t)row * DMODEL + d];
      feat[i * DMODEL + d] = t; ls += t; ls2 += t * t;
    }
    float mean = blk_sum(ls, red) * (1.f / DMODEL);
    float var  = blk_sum(ls2, red) * (1.f / DMODEL) - mean * mean;
    float rs = rsqrtf(var + 1e-5f);
    __syncthreads();
    for (int d = tid; d < DMODEL; d += 256)
      feat[i * DMODEL + d] = (feat[i * DMODEL + d] - mean) * rs * ng[d] + nb[d];
    __syncthreads();
  }
  for (int idx = tid; idx < BAGSZ * NREL; idx += 256) {
    int i = idx / NREL, j = idx % NREL;
    float acc = bp[j];
    for (int d = 0; d < DMODEL; ++d) acc += feat[i * DMODEL + d] * Wp[(size_t)d * NREL + j];
    out[NREL + idx] = acc;
  }
  __syncthreads();
  for (int j = tid; j < NREL; j += 256) {
    float m = out[NREL + j];
    for (int i = 1; i < BAGSZ; ++i) m = fmaxf(m, out[NREL + i * NREL + j]);
    out[j] = m;
  }
}

// ---------------------------------------------------------------------------
// Host driver
// ---------------------------------------------------------------------------
extern "C" void kernel_launch(void* const* d_in, const int* in_sizes, int n_in,
                              void* d_out, int out_size, void* d_ws, size_t ws_size,
                              hipStream_t stream) {
  (void)in_sizes; (void)n_in; (void)out_size; (void)ws_size;
  const float* emb   = (const float*)d_in[0];
  const int*   spans = (const int*)d_in[1];
  const float* Wu = (const float*)d_in[2],  *bu  = (const float*)d_in[3];
  const float* Wv = (const float*)d_in[4],  *bv  = (const float*)d_in[5];
  const float* Wl = (const float*)d_in[6],  *bl  = (const float*)d_in[7];
  const float* Wq = (const float*)d_in[8],  *bq  = (const float*)d_in[9];
  const float* Wk = (const float*)d_in[10], *bk  = (const float*)d_in[11];
  const float* Wvm= (const float*)d_in[12], *bvm = (const float*)d_in[13];
  const float* Wo = (const float*)d_in[14], *bo  = (const float*)d_in[15];
  const float* F1 = (const float*)d_in[16], *f1  = (const float*)d_in[17];
  const float* F2 = (const float*)d_in[18], *f2  = (const float*)d_in[19];
  const float* g1 = (const float*)d_in[20], *be1 = (const float*)d_in[21];
  const float* g2 = (const float*)d_in[22], *be2 = (const float*)d_in[23];
  const float* ng = (const float*)d_in[24], *nb  = (const float*)d_in[25];
  const float* Wp = (const float*)d_in[26], *bp  = (const float*)d_in[27];
  float* out = (float*)d_out;

  // ---- workspace bump allocator (256B aligned regions) ----
  char* wsp = (char*)d_ws;
  auto alloc = [&](size_t bytes) -> void* {
    void* p = (void*)wsp;
    wsp += (bytes + 255) & ~(size_t)255;
    return p;
  };
  const size_t DD  = (size_t)DMODEL * DMODEL;
  const size_t DFF = (size_t)DMODEL * DFFN;
  // transposed bf16 weights ([N][K] layout for the GEMM's B operand)
  unsigned short* wWuT  = (unsigned short*)alloc(DD * 2);
  unsigned short* wWvT  = (unsigned short*)alloc(DD * 2);
  unsigned short* wWlT  = (unsigned short*)alloc(DD * 2);
  unsigned short* wWqT  = (unsigned short*)alloc(NLAYER * DD * 2);
  unsigned short* wWkT  = (unsigned short*)alloc(NLAYER * DD * 2);
  unsigned short* wWvmT = (unsigned short*)alloc(NLAYER * DD * 2);
  unsigned short* wWoT  = (unsigned short*)alloc(NLAYER * DD * 2);
  unsigned short* wF1T  = (unsigned short*)alloc(NLAYER * DFF * 2);   // [1024,768] per layer
  unsigned short* wF2T  = (unsigned short*)alloc(NLAYER * DFF * 2);   // [768,1024] per layer
  float*          htbF  = (float*)alloc((size_t)NNODES * DMODEL * 4);
  unsigned short* htbB  = (unsigned short*)alloc((size_t)BM * DMODEL * 2);      // padded to 128 rows
  float*          uF    = (float*)alloc((size_t)NNODES * DMODEL * 4);
  float*          vF    = (float*)alloc((size_t)NNODES * DMODEL * 4);
  unsigned short* pairA = (unsigned short*)alloc((size_t)TCELLS * DMODEL * 2);
  float*          xF    = (float*)alloc((size_t)TCELLS * DMODEL * 4);
  unsigned short* xB    = (unsigned short*)alloc((size_t)TCELLS * DMODEL * 2);
  unsigned short* qB    = (unsigned short*)alloc((size_t)TCELLS * DMODEL * 2);
  unsigned short* kB    = (unsigned short*)alloc((size_t)TCELLS * DMODEL * 2);
  unsigned short* vT    = (unsigned short*)alloc((size_t)(DMODEL + BM) * TCELLS * 2); // V^T, padded rows
  float*          Sbuf  = (float*)alloc((size_t)TCELLS * TCELLS * 4);
  unsigned short* Pbuf  = (unsigned short*)alloc((size_t)TCELLS * TCELLS * 2);
  unsigned short* attnO = (unsigned short*)alloc((size_t)TCELLS * DMODEL * 2);
  float*          tmpF  = (float*)alloc((size_t)TCELLS * DMODEL * 4);
  unsigned short* midB  = (unsigned short*)alloc((size_t)TCELLS * DFFN * 2);

  auto convT = [&](const float* s, unsigned short* d, int K_, int N_) {
    dim3 g((N_ + 31) / 32, (K_ + 31) / 32);
    k_f32_to_bf16_t<<<g, dim3(256), 0, stream>>>(s, d, K_, N_);
  };
  auto gemm = [&](int M, int N, int K, const unsigned short* A, int lda,
                  const unsigned short* B, int ldb, const float* bias,
                  float* oF, unsigned short* oB, unsigned short* oBT,
                  int ldc, int ldct, int relu) {
    dim3 g((N + BN - 1) / BN, (M + BM - 1) / BM);
    gemm_bf16_wmma<<<g, dim3(256), 0, stream>>>(M, N, K, A, lda, B, ldb,
                                                bias, oF, oB, oBT, ldc, ldct, relu);
  };

  // 1) weights -> bf16, transposed to [N][K]
  convT(Wu, wWuT, DMODEL, DMODEL);
  convT(Wv, wWvT, DMODEL, DMODEL);
  convT(Wl, wWlT, DMODEL, DMODEL);
  for (int l = 0; l < NLAYER; ++l) {
    convT(Wq  + l * DD,  wWqT  + l * DD,  DMODEL, DMODEL);
    convT(Wk  + l * DD,  wWkT  + l * DD,  DMODEL, DMODEL);
    convT(Wvm + l * DD,  wWvmT + l * DD,  DMODEL, DMODEL);
    convT(Wo  + l * DD,  wWoT  + l * DD,  DMODEL, DMODEL);
    convT(F1  + l * DFF, wF1T  + l * DFF, DMODEL, DFFN);   // -> [1024,768]
    convT(F2  + l * DFF, wF2T  + l * DFF, DFFN, DMODEL);   // -> [768,1024]
  }

  // 2) span max-pool -> node embeddings
  k_span_pool<<<dim3(NNODES), dim3(256), 0, stream>>>(emb, spans, htbF, htbB);

  // 3) u = htb@Wu+bu, v = htb@Wv+bv
  gemm(NNODES, DMODEL, DMODEL, htbB, DMODEL, wWuT, DMODEL, bu, uF, nullptr, nullptr, DMODEL, 0, 0);
  gemm(NNODES, DMODEL, DMODEL, htbB, DMODEL, wWvT, DMODEL, bv, vF, nullptr, nullptr, DMODEL, 0, 0);

  // 4) pairA = relu(u_i + v_j);  x = relu(pairA@Wl + bl)
  k_pair<<<dim3(TCELLS), dim3(256), 0, stream>>>(uF, vF, pairA);
  gemm(TCELLS, DMODEL, DMODEL, pairA, DMODEL, wWlT, DMODEL, bl, xF, xB, nullptr, DMODEL, 0, 1);

  // 5) MatTransformer layers
  const float scale = 0.10206207261596575f;   // 1/sqrt(96)
  for (int l = 0; l < NLAYER; ++l) {
    gemm(TCELLS, DMODEL, DMODEL, xB, DMODEL, wWqT  + (size_t)l * DD, DMODEL,
         bq  + l * DMODEL, nullptr, qB, nullptr, DMODEL, 0, 0);
    gemm(TCELLS, DMODEL, DMODEL, xB, DMODEL, wWkT  + (size_t)l * DD, DMODEL,
         bk  + l * DMODEL, nullptr, kB, nullptr, DMODEL, 0, 0);
    // V projection stored TRANSPOSED: vT[dim][token]
    gemm(TCELLS, DMODEL, DMODEL, xB, DMODEL, wWvmT + (size_t)l * DD, DMODEL,
         bvm + l * DMODEL, nullptr, nullptr, vT, DMODEL, TCELLS, 0);

    for (int h = 0; h < NHEAD; ++h) {
      // S = Q_h @ K_h^T   (B = K rows [token][DK] = [N,K] with ld DMODEL)
      gemm(TCELLS, TCELLS, DHEAD, qB + h * DHEAD, DMODEL,
           kB + h * DHEAD, DMODEL, nullptr, Sbuf, nullptr, nullptr, TCELLS, 0, 0);
      k_softmax<<<dim3(TCELLS), dim3(256), 0, stream>>>(Sbuf, Pbuf, TCELLS, scale);
      // attnO[:, h*96:] = P @ V_h   (B = V^T rows [dim][token] = [N,K] with ld TCELLS)
      gemm(TCELLS, DHEAD, TCELLS, Pbuf, TCELLS,
           vT + (size_t)(h * DHEAD) * TCELLS, TCELLS,
           nullptr, nullptr, attnO + h * DHEAD, nullptr, DMODEL, 0, 0);
    }

    // O-projection, residual + LN1
    gemm(TCELLS, DMODEL, DMODEL, attnO, DMODEL, wWoT + (size_t)l * DD, DMODEL,
         bo + l * DMODEL, tmpF, nullptr, nullptr, DMODEL, 0, 0);
    k_add_ln<<<dim3(TCELLS), dim3(256), 0, stream>>>(xF, tmpF, g1 + l * DMODEL,
                                                     be1 + l * DMODEL, xF, xB);
    // FFN, residual + LN2
    gemm(TCELLS, DFFN, DMODEL, xB, DMODEL, wF1T + (size_t)l * DFF, DMODEL,
         f1 + l * DFFN, nullptr, midB, nullptr, DFFN, 0, 1);
    gemm(TCELLS, DMODEL, DFFN, midB, DFFN, wF2T + (size_t)l * DFF, DFFN,
         f2 + l * DMODEL, tmpF, nullptr, nullptr, DMODEL, 0, 0);
    k_add_ln<<<dim3(TCELLS), dim3(256), 0, stream>>>(xF, tmpF, g2 + l * DMODEL,
                                                     be2 + l * DMODEL, xF, xB);
  }

  // 6) gather + LN + predictor + bag max
  k_final<<<dim3(1), dim3(256), 0, stream>>>(xF, ng, nb, Wp, bp, out);
}
